// EGAN_Layer_38448547234242
// MI455X (gfx1250) — matrix operand order repaired and checked by reading, not compiled, not run on hardware
//
#include <hip/hip_runtime.h>
#include <hip/hip_bf16.h>

typedef __bf16 bf16_t;
typedef __attribute__((ext_vector_type(16))) __bf16 v16bf;
typedef __attribute__((ext_vector_type(8)))  __bf16 v8bf;
typedef __attribute__((ext_vector_type(8)))  float  v8f;

__device__ __forceinline__ v8f wmma_bf16f32(v16bf a, v16bf b, v8f c) {
  // D = A(16x32 bf16) x B(32x16 bf16) + C(16x16 f32)
  return __builtin_amdgcn_wmma_f32_16x16x32_bf16(false, a, false, b, (short)0, c,
                                                 false, false);
}

// A-operand storage permutation: identity, so the per-lane WMMA A fragment
// (k = half*16 + j) is the contiguous 16-element run at offset half*16 within
// each 32-block, matching the B operand. Involutive.
__device__ __host__ __forceinline__ int permA(int c) {
  return c;
}

// ---------------------------------------------------------------------------
// Weight conversion: fp32 [K, Ncols] row-major  ->  bf16 BT [Ncols, K]
// (one-time; makes the WMMA B operand a contiguous per-lane load)
// ---------------------------------------------------------------------------
__global__ void __launch_bounds__(256) egan_cvt_transpose(
    const float* __restrict__ in, bf16_t* __restrict__ out, int K, int Ncols) {
  const int total = K * Ncols;
  for (int i = blockIdx.x * blockDim.x + threadIdx.x; i < total;
       i += gridDim.x * blockDim.x) {
    const int n = i / K;
    const int k = i - n * K;
    out[i] = (bf16_t)in[(size_t)k * Ncols + n];
  }
}

// ---------------------------------------------------------------------------
// Per-node prep: g_src = (equ@W_equ)*16 ; h_src = h*16 ;
// g2p = g_src@W_gproj ; gram[a,b] = sum_i g2p[i,a]*g2p[i,b]
// bf16 activation outputs are stored A-permuted. gsrc_bi is [i][node][e].
// ---------------------------------------------------------------------------
__global__ void __launch_bounds__(128) egan_prep(
    const float* __restrict__ equ, const float* __restrict__ h,
    const float* __restrict__ W_equ, const float* __restrict__ W_gproj,
    float* __restrict__ gsrc_f, bf16_t* __restrict__ gsrc_bi,
    float* __restrict__ hsrc_f, bf16_t* __restrict__ h2_b,
    bf16_t* __restrict__ gram_b, int N) {
  __shared__ float equ_s[48];
  __shared__ float gsrc_s[768];
  __shared__ float g2p_s[96];
  const int node = blockIdx.x;
  const int t = threadIdx.x;

  if (t < 48) equ_s[t] = equ[(size_t)node * 48 + t];
  __syncthreads();

  for (int e = t; e < 256; e += 128) {
    const int pe = permA(e);
#pragma unroll
    for (int i = 0; i < 3; ++i) {
      float g = 0.f;
#pragma unroll
      for (int m = 0; m < 16; ++m) g += equ_s[i * 16 + m] * W_equ[m * 256 + e];
      g *= 16.0f;  // sqrt(E)
      gsrc_s[i * 256 + e] = g;
      gsrc_f[(size_t)node * 768 + i * 256 + e] = g;                  // logical
      gsrc_bi[(size_t)i * N * 256 + (size_t)node * 256 + pe] = (bf16_t)g;
    }
    float hv = h[(size_t)node * 256 + e] * 16.0f;
    hsrc_f[(size_t)node * 256 + e] = hv;
    h2_b[(size_t)node * 512 + 256 + pe] = (bf16_t)hv;  // concat second half
  }
  __syncthreads();

  if (t < 96) {
    const int i = t >> 5, a = t & 31;
    float acc = 0.f;
    for (int e = 0; e < 256; ++e) acc += gsrc_s[i * 256 + e] * W_gproj[e * 32 + a];
    g2p_s[t] = acc;
  }
  __syncthreads();

  for (int idx = t; idx < 1024; idx += 128) {
    const int a = idx >> 5, b = idx & 31;
    float gr = g2p_s[a] * g2p_s[b] + g2p_s[32 + a] * g2p_s[32 + b] +
               g2p_s[64 + a] * g2p_s[64 + b];
    gram_b[(size_t)node * 1024 + permA(idx)] = (bf16_t)gr;
  }
}

// ---------------------------------------------------------------------------
// bf16 WMMA GEMM: C[M,Ncols] = act((A[M,K]@B + bias)*scale) + R
// A is A-permuted bf16 [M,K]; B passed transposed BT[Ncols,K].
// One 16x16 tile per wave, 8 waves/block. Per k-step: 1 contiguous 32B load
// per operand + 1 v_wmma.
// bmode: 0 = linear bf16 store, 1 = A-permuted bf16 store (feeds next GEMM's A),
//        2 = transposed bf16 store Cb[col*ldcb + row] (feeds attention B side).
// ---------------------------------------------------------------------------
__global__ void __launch_bounds__(256) egan_gemm_bf16(
    const bf16_t* __restrict__ A, int K, int lda,
    const bf16_t* __restrict__ BT,
    const float* __restrict__ bias,
    const float* __restrict__ residual, int ldr,
    float scale, int do_relu,
    bf16_t* __restrict__ Cb, int bmode, int ldcb,
    float* __restrict__ Cf, int ldcf, int Mrows) {
  const int lane = threadIdx.x & 31;
  const int wave = threadIdx.x >> 5;
  const int kh = lane >> 4;
  const int lr = lane & 15;
  const int row0 = blockIdx.y * 128 + wave * 16;
  const int col0 = blockIdx.x * 16;
  if (row0 >= Mrows) return;

  const bf16_t* aptr = A + (size_t)(row0 + lr) * lda + kh * 16;
  const bf16_t* bptr = BT + (size_t)(col0 + lr) * K + kh * 16;

  v8f acc = {0.f, 0.f, 0.f, 0.f, 0.f, 0.f, 0.f, 0.f};
#pragma unroll 4
  for (int k0 = 0; k0 < K; k0 += 32) {
    v16bf a = *(const v16bf*)(aptr + k0);
    v16bf b = *(const v16bf*)(bptr + k0);
    acc = wmma_bf16f32(a, b, acc);
  }

  // C layout: vgpr r -> row r + 8*kh, col lr
  const int col = col0 + lr;
  float vals[8];
#pragma unroll
  for (int r = 0; r < 8; ++r) {
    float v = acc[r];
    if (bias) v += bias[col];
    v *= scale;
    if (do_relu) v = v > 0.f ? v : 0.f;
    if (residual) v += residual[(size_t)(row0 + r + 8 * kh) * ldr + col];
    vals[r] = v;
  }
  if (Cf) {
#pragma unroll
    for (int r = 0; r < 8; ++r)
      Cf[(size_t)(row0 + r + 8 * kh) * ldcf + col] = vals[r];
  }
  if (Cb) {
    if (bmode == 2) {  // packed 16B transposed store
      v8bf pack;
#pragma unroll
      for (int r = 0; r < 8; ++r) pack[r] = (bf16_t)vals[r];
      *(v8bf*)(Cb + (size_t)col * ldcb + row0 + 8 * kh) = pack;
    } else {
      const int scol = (bmode == 1) ? permA(col) : col;
#pragma unroll
      for (int r = 0; r < 8; ++r)
        Cb[(size_t)(row0 + r + 8 * kh) * ldcb + scol] = (bf16_t)vals[r];
    }
  }
}

// ---------------------------------------------------------------------------
// Flash attention (one 16-query tile per wave, 8 waves/block, grid.y = head).
// Q: A-permuted [N,256]; K: linear [N,256]; VT: [256,N]; VGT: [256,3N]
// (VGT[d][i*N+node]). Outputs OutV [N,256], OutG [3N,256], A-permuted bf16.
// ---------------------------------------------------------------------------
__global__ void __launch_bounds__(256) egan_flash_attn(
    const bf16_t* __restrict__ Qm, const bf16_t* __restrict__ Km,
    const bf16_t* __restrict__ VT, const bf16_t* __restrict__ VGT,
    bf16_t* __restrict__ OutV, bf16_t* __restrict__ OutG, int N) {
  __shared__ __align__(32) bf16_t plds[8][16][32];
  const int lane = threadIdx.x & 31;
  const int wave = threadIdx.x >> 5;
  const int kh = lane >> 4;
  const int lr = lane & 15;
  const int head = blockIdx.y;
  const int hoff = head * 32;
  const int q0 = blockIdx.x * 128 + wave * 16;

  // Q tile (pre-scaled by D^-0.5), A-permuted -> contiguous fragment load
  v16bf qa = *(const v16bf*)(Qm + (size_t)(q0 + lr) * 256 + hoff + kh * 16);

  const v8f zero8 = {0.f, 0.f, 0.f, 0.f, 0.f, 0.f, 0.f, 0.f};
  v8f ov0 = zero8, ov1 = zero8;
  v8f og[6];
#pragma unroll
  for (int i = 0; i < 6; ++i) og[i] = zero8;
  float mi[8], li[8];
#pragma unroll
  for (int r = 0; r < 8; ++r) { mi[r] = -1e30f; li[r] = 0.f; }

  const int p0c = permA(lr), p1c = permA(16 + lr);

  for (int kb = 0; kb < N; kb += 32) {
    // S = Q K^T : B column j = key row (kb[+16]+lr), dims contiguous
    v8f s0, s1;
    {
      v16bf bk = *(const v16bf*)(Km + (size_t)(kb + lr) * 256 + hoff + kh * 16);
      s0 = wmma_bf16f32(qa, bk, zero8);
      bk = *(const v16bf*)(Km + (size_t)(kb + 16 + lr) * 256 + hoff + kh * 16);
      s1 = wmma_bf16f32(qa, bk, zero8);
    }
    // online softmax; each C row lives in one 16-lane half
    float pm[8];
#pragma unroll
    for (int r = 0; r < 8; ++r) pm[r] = fmaxf(s0[r], s1[r]);
#pragma unroll
    for (int m = 1; m <= 8; m <<= 1)
#pragma unroll
      for (int r = 0; r < 8; ++r) pm[r] = fmaxf(pm[r], __shfl_xor(pm[r], m, 32));

    float sf[8], rs[8], p0[8], p1[8];
#pragma unroll
    for (int r = 0; r < 8; ++r) {
      float mn = fmaxf(mi[r], pm[r]);
      sf[r] = __expf(mi[r] - mn);
      p0[r] = __expf(s0[r] - mn);
      p1[r] = __expf(s1[r] - mn);
      rs[r] = p0[r] + p1[r];
      mi[r] = mn;
    }
#pragma unroll
    for (int m = 1; m <= 8; m <<= 1)
#pragma unroll
      for (int r = 0; r < 8; ++r) rs[r] += __shfl_xor(rs[r], m, 32);
#pragma unroll
    for (int r = 0; r < 8; ++r) {
      li[r] = li[r] * sf[r] + rs[r];
      ov0[r] *= sf[r];
      ov1[r] *= sf[r];
#pragma unroll
      for (int c = 0; c < 6; ++c) og[c][r] *= sf[r];
    }

    // C->A transpose of P via per-wave LDS tile; write side applies permA so
    // the read side is one contiguous 32B fragment load.
    __syncthreads();
#pragma unroll
    for (int r = 0; r < 8; ++r) {
      plds[wave][r + 8 * kh][p0c] = (bf16_t)p0[r];
      plds[wave][r + 8 * kh][p1c] = (bf16_t)p1[r];
    }
    __syncthreads();
    v16bf pa = *(const v16bf*)(&plds[wave][lr][kh * 16]);

    // out_v += P @ V : VT rows are d, contiguous over keys
#pragma unroll
    for (int c = 0; c < 2; ++c) {
      v16bf bv = *(const v16bf*)(VT + (size_t)(hoff + c * 16 + lr) * N + kb + kh * 16);
      if (c == 0) ov0 = wmma_bf16f32(pa, bv, ov0);
      else        ov1 = wmma_bf16f32(pa, bv, ov1);
    }
    // out_g += P @ VG for the 3 vector channels
#pragma unroll
    for (int vc = 0; vc < 3; ++vc)
#pragma unroll
      for (int c = 0; c < 2; ++c) {
        v16bf bg = *(const v16bf*)(VGT + (size_t)(hoff + c * 16 + lr) * 3 * N +
                                   (size_t)vc * N + kb + kh * 16);
        og[vc * 2 + c] = wmma_bf16f32(pa, bg, og[vc * 2 + c]);
      }
  }

#pragma unroll
  for (int r = 0; r < 8; ++r) {
    const float inv = 1.f / li[r];
    const int row = q0 + r + 8 * kh;
    OutV[(size_t)row * 256 + permA(hoff + lr)] = (bf16_t)(ov0[r] * inv);
    OutV[(size_t)row * 256 + permA(hoff + 16 + lr)] = (bf16_t)(ov1[r] * inv);
#pragma unroll
    for (int vc = 0; vc < 3; ++vc)
#pragma unroll
      for (int c = 0; c < 2; ++c)
        OutG[((size_t)row * 3 + vc) * 256 + permA(hoff + c * 16 + lr)] =
            (bf16_t)(og[vc * 2 + c][r] * inv);
  }
}

// ---------------------------------------------------------------------------
extern "C" void kernel_launch(void* const* d_in, const int* in_sizes, int n_in,
                              void* d_out, int out_size, void* d_ws, size_t ws_size,
                              hipStream_t stream) {
  const float* equ    = (const float*)d_in[0];
  const float* h      = (const float*)d_in[1];
  // d_in[2] edge_index, d_in[3] edge_fea: dead args in the reference
  const float* W_equ  = (const float*)d_in[4];
  const float* W_gproj= (const float*)d_in[5];
  const float* W_vg   = (const float*)d_in[6];
  const float* W_g1   = (const float*)d_in[7];
  const float* b_g1   = (const float*)d_in[8];
  const float* W_g2   = (const float*)d_in[9];
  const float* b_g2   = (const float*)d_in[10];
  const float* W_q    = (const float*)d_in[11];
  const float* b_q    = (const float*)d_in[12];
  const float* W_k    = (const float*)d_in[13];
  const float* b_k    = (const float*)d_in[14];
  const float* W_v    = (const float*)d_in[15];
  const float* b_v    = (const float*)d_in[16];
  const float* W_ng   = (const float*)d_in[17];
  const float* b_ng   = (const float*)d_in[18];
  const float* W_gout = (const float*)d_in[19];
  const float* W_gdec = (const float*)d_in[20];
  const float* W_hdec = (const float*)d_in[21];
  const float* b_hdec = (const float*)d_in[22];
  float* out = (float*)d_out;

  const int N = in_sizes[1] / 256;  // 4096
  const float scaling = 0.17677669529663689f;  // 32^-0.5

  char* ws = (char*)d_ws;
  size_t off = 0;
  auto alloc = [&](size_t bytes) -> void* {
    void* p = ws + off;
    off = (off + bytes + 255) & ~(size_t)255;
    return p;
  };

  // transposed bf16 weights BT[Ncols][K]
  bf16_t* wg1_t   = (bf16_t*)alloc((size_t)1024 * 512 * 2);
  bf16_t* wg2_t   = (bf16_t*)alloc((size_t)512 * 256 * 2);
  bf16_t* wq_t    = (bf16_t*)alloc((size_t)512 * 256 * 2);
  bf16_t* wk_t    = (bf16_t*)alloc((size_t)512 * 256 * 2);
  bf16_t* wv_t    = (bf16_t*)alloc((size_t)512 * 256 * 2);
  bf16_t* wvg_t   = (bf16_t*)alloc((size_t)256 * 256 * 2);
  bf16_t* wng_t   = (bf16_t*)alloc((size_t)256 * 256 * 2);
  bf16_t* wgout_t = (bf16_t*)alloc((size_t)256 * 256 * 2);
  bf16_t* wgdec_t = (bf16_t*)alloc((size_t)256 * 16 * 2);
  bf16_t* whdec_t = (bf16_t*)alloc((size_t)256 * 256 * 2);
  // activations
  float*  gsrc_f  = (float*) alloc((size_t)N * 768 * 4);  // [node][i][e]
  bf16_t* gsrc_bi = (bf16_t*)alloc((size_t)N * 768 * 2);  // [i][node][e] permA
  float*  hsrc_f  = (float*) alloc((size_t)N * 256 * 4);
  bf16_t* gram_b  = (bf16_t*)alloc((size_t)N * 1024 * 2); // permA
  bf16_t* hid_b   = (bf16_t*)alloc((size_t)N * 512 * 2);  // permA
  bf16_t* h2_b    = (bf16_t*)alloc((size_t)N * 512 * 2);  // permA
  bf16_t* q_b     = (bf16_t*)alloc((size_t)N * 256 * 2);  // permA
  bf16_t* k_b     = (bf16_t*)alloc((size_t)N * 256 * 2);  // linear
  bf16_t* vt_b    = (bf16_t*)alloc((size_t)N * 256 * 2);  // [256][N]
  bf16_t* vgt_b   = (bf16_t*)alloc((size_t)N * 768 * 2);  // [256][3N]
  bf16_t* ov_b    = (bf16_t*)alloc((size_t)N * 256 * 2);  // permA
  bf16_t* og_b    = (bf16_t*)alloc((size_t)N * 768 * 2);  // permA
  bf16_t* htot_b  = (bf16_t*)alloc((size_t)N * 256 * 2);  // permA
  bf16_t* gtot_b  = (bf16_t*)alloc((size_t)N * 768 * 2);  // permA

  auto cvt = [&](const float* src, bf16_t* dst, int K, int Ncols) {
    egan_cvt_transpose<<<dim3(256), dim3(256), 0, stream>>>(src, dst, K, Ncols);
  };
  cvt(W_g1, wg1_t, 1024, 512);
  cvt(W_g2, wg2_t, 512, 256);
  cvt(W_q, wq_t, 512, 256);
  cvt(W_k, wk_t, 512, 256);
  cvt(W_v, wv_t, 512, 256);
  cvt(W_vg, wvg_t, 256, 256);
  cvt(W_ng, wng_t, 256, 256);
  cvt(W_gout, wgout_t, 256, 256);
  cvt(W_gdec, wgdec_t, 256, 16);
  cvt(W_hdec, whdec_t, 256, 256);

  egan_prep<<<dim3(N), dim3(128), 0, stream>>>(equ, h, W_equ, W_gproj, gsrc_f,
                                               gsrc_bi, hsrc_f, h2_b, gram_b, N);

  auto gemm = [&](const bf16_t* A, int K, int lda, const bf16_t* BT,
                  const float* bias, const float* resid, int ldr, float scale,
                  int relu, bf16_t* Cb, int bmode, int ldcb, float* Cf, int ldcf,
                  int M, int Ncols) {
    dim3 grid(Ncols / 16, (M + 127) / 128);
    egan_gemm_bf16<<<grid, 256, 0, stream>>>(A, K, lda, BT, bias, resid, ldr,
                                             scale, relu, Cb, bmode, ldcb, Cf,
                                             ldcf, M);
  };

  // gram MLP
  gemm(gram_b, 1024, 1024, wg1_t, b_g1, nullptr, 0, 1.f, 1, hid_b, 1, 512,
       nullptr, 0, N, 512);
  gemm(hid_b, 512, 512, wg2_t, b_g2, nullptr, 0, 1.f, 0, h2_b, 1, 512,
       nullptr, 0, N, 256);
  // Q (pre-scaled, permA), K (linear), V (transposed) from h2 = [g2 | h_src]
  gemm(h2_b, 512, 512, wq_t, b_q, nullptr, 0, scaling, 0, q_b, 1, 256,
       nullptr, 0, N, 256);
  gemm(h2_b, 512, 512, wk_t, b_k, nullptr, 0, 1.f, 0, k_b, 0, 256,
       nullptr, 0, N, 256);
  gemm(h2_b, 512, 512, wv_t, b_v, nullptr, 0, 1.f, 0, vt_b, 2, N,
       nullptr, 0, N, 256);
  // vgT = (g_src @ W_vg)^T, rows ordered [i][node] -> VGT[d][i*N+node]
  gemm(gsrc_bi, 256, 256, wvg_t, nullptr, nullptr, 0, 1.f, 0, vgt_b, 2, 3 * N,
       nullptr, 0, 3 * N, 256);

  // fused flash attention over all heads
  egan_flash_attn<<<dim3(N / 128, 8), 256, 0, stream>>>(q_b, k_b, vt_b, vgt_b,
                                                        ov_b, og_b, N);

  // h_total = h_src + (out_v @ W_ng + b_ng)
  gemm(ov_b, 256, 256, wng_t, b_ng, hsrc_f, 256, 1.f, 0, htot_b, 1, 256,
       nullptr, 0, N, 256);
  // h_out = h_total @ W_hdec + b_hdec  -> second output block
  gemm(htot_b, 256, 256, whdec_t, b_hdec, nullptr, 0, 1.f, 0, nullptr, 0, 0,
       out + (size_t)N * 48, 256, N, 256);
  // g_total = g_src + out_g @ W_gout   (rows node*3+vc)
  gemm(og_b, 256, 256, wgout_t, nullptr, gsrc_f, 256, 1.f, 0, gtot_b, 1, 256,
       nullptr, 0, 3 * N, 256);
  // equ_out = g_total @ W_gdec -> first output block [N,3,16]
  gemm(gtot_b, 256, 256, wgdec_t, nullptr, nullptr, 0, 1.f, 0, nullptr, 0, 0,
       out, 16, 3 * N, 16);
}
